// MultiHeadAttention_35751307772605
// MI455X (gfx1250) — matrix operand order repaired and checked by reading, not compiled, not run on hardware
//
#include <hip/hip_runtime.h>

typedef __bf16 bf16;
typedef __attribute__((ext_vector_type(16))) __bf16 v16bf;
typedef __attribute__((ext_vector_type(8)))  __bf16 v8bf;
typedef __attribute__((ext_vector_type(8)))  float   v8f;

#define D_MODEL 1024
#define NUM_HEADS 16
#define DEPTH 64
#define SEQ 2048
#define BATCH 2
#define ROWS (BATCH * SEQ) /* 4096 */

#define BM 128
#define BN 64
#define BK 32
#define LDT (BK + 8)  /* 80B row stride, keeps 16B alignment */
#define AW 8          /* attention waves (q-tiles) per block */

__device__ __forceinline__ v8f wmma_bf16(v16bf a, v16bf b, v8f c) {
  // D = A(16x32 bf16) * B(32x16 bf16) + C(16x16 f32)
  return __builtin_amdgcn_wmma_f32_16x16x32_bf16(false, a, false, b, (short)0, c,
                                                 false, false);
}

// A-fragment: 16x32 bf16 tile from row-major storage (lane = row L&15).
// half0 lanes hold k=0..7 & 16..23; half1 lanes hold k=8..15 & 24..31.
__device__ __forceinline__ v16bf load_frag_a(const bf16* base, int stride) {
  const int lane = threadIdx.x & 31;
  const bf16* p = base + (size_t)(lane & 15) * stride + (lane >> 4) * 8;
  union { v16bf v; v8bf h[2]; } u;
  u.h[0] = *(const v8bf*)(p);
  u.h[1] = *(const v8bf*)(p + 16);
  return u.v;
}

// B-fragment: 32x16 bf16 tile stored transposed row-major [n][k] (lane = col L&15).
// half0 lanes hold k=0..7 & 16..23; half1 lanes hold k=8..15 & 24..31.
__device__ __forceinline__ v16bf load_frag_b(const bf16* base, int stride) {
  const int lane = threadIdx.x & 31;
  const bf16* p = base + (size_t)(lane & 15) * stride + (lane >> 4) * 8;
  union { v16bf v; v8bf h[2]; } u;
  u.h[0] = *(const v8bf*)(p);
  u.h[1] = *(const v8bf*)(p + 16);
  return u.v;
}

__global__ void k_cvt(const float* __restrict__ in, bf16* __restrict__ out, int n) {
  int i = blockIdx.x * 256 + threadIdx.x;
  if (i < n) out[i] = (bf16)in[i];
}

// ---- pipelined GEMM core helpers (256 threads, BM=128 x BN=64 tile) ----
__device__ __forceinline__ void gemm_stage(const bf16* __restrict__ X,
                                           const bf16* __restrict__ W,
                                           int rowBase, int colBase, int kk, int t,
                                           unsigned int* ra, unsigned int* rb) {
#pragma unroll
  for (int i = 0; i < 8; ++i) {          // A tile 128x32 as dwords
    int idx = i * 256 + t;
    int r = idx >> 4;
    int c = (idx & 15) * 2;
    ra[i] = *(const unsigned int*)&X[(size_t)(rowBase + r) * D_MODEL + kk + c];
  }
#pragma unroll
  for (int i = 0; i < 4; ++i) {          // B tile 32x64 as dwords
    int idx = i * 256 + t;
    int k = idx >> 5;
    int n = (idx & 31) * 2;
    rb[i] = *(const unsigned int*)&W[(size_t)(kk + k) * D_MODEL + colBase + n];
  }
}

__device__ __forceinline__ void gemm_commit(bf16 (*As)[LDT], bf16 (*Bs)[LDT], int t,
                                            const unsigned int* ra,
                                            const unsigned int* rb) {
#pragma unroll
  for (int i = 0; i < 8; ++i) {
    int idx = i * 256 + t;
    int r = idx >> 4;
    int c = (idx & 15) * 2;
    *(unsigned int*)&As[r][c] = ra[i];
  }
#pragma unroll
  for (int i = 0; i < 4; ++i) {
    int idx = i * 256 + t;
    int k = idx >> 5;
    int n = (idx & 31) * 2;
    union { unsigned int u; bf16 b[2]; } w;
    w.u = rb[i];
    Bs[n][k] = w.b[0];       // transposed store: Bs[n][k] = W[kk+k][colBase+n]
    Bs[n + 1][k] = w.b[1];
  }
}

// ---------------- QKV projection: [4096,1024] x [1024,1024] (x3 via grid.z) -------
__global__ __launch_bounds__(256)
void k_gemm_qkv(const bf16* __restrict__ X,
                const bf16* __restrict__ Wq, const bf16* __restrict__ Wk,
                const bf16* __restrict__ Wv,
                const float* __restrict__ bq, const float* __restrict__ bk,
                const float* __restrict__ bv,
                bf16* __restrict__ Q, bf16* __restrict__ Ko, bf16* __restrict__ Vt) {
  __shared__ __align__(16) bf16 As[2][BM][LDT];
  __shared__ __align__(16) bf16 Bs[2][BN][LDT];
  const int z = blockIdx.z;
  const bf16* W = (z == 0) ? Wq : (z == 1) ? Wk : Wv;
  const float* bias = (z == 0) ? bq : (z == 1) ? bk : bv;
  const int rowBase = blockIdx.y * BM;
  const int colBase = blockIdx.x * BN;
  const int t = threadIdx.x;
  const int wid = t >> 5;
  const int lane = t & 31;
  const int half = lane >> 4;

  v8f acc[4] = {};
  unsigned int ra[8], rb[4];

  gemm_stage(X, W, rowBase, colBase, 0, t, ra, rb);
  gemm_commit(As[0], Bs[0], t, ra, rb);

  const int NIT = D_MODEL / BK;  // 32
  for (int it = 0; it < NIT; ++it) {
    const int buf = it & 1;
    __syncthreads();
    const bool pre = (it + 1 < NIT);
    if (pre) gemm_stage(X, W, rowBase, colBase, (it + 1) * BK, t, ra, rb);

    v16bf a = load_frag_a(&As[buf][wid * 16][0], LDT);
    v16bf b0 = load_frag_b(&Bs[buf][0][0], LDT);
    v16bf b1 = load_frag_b(&Bs[buf][16][0], LDT);
    v16bf b2 = load_frag_b(&Bs[buf][32][0], LDT);
    v16bf b3 = load_frag_b(&Bs[buf][48][0], LDT);
    acc[0] = wmma_bf16(a, b0, acc[0]);
    acc[1] = wmma_bf16(a, b1, acc[1]);
    acc[2] = wmma_bf16(a, b2, acc[2]);
    acc[3] = wmma_bf16(a, b3, acc[3]);

    if (pre) gemm_commit(As[buf ^ 1], Bs[buf ^ 1], t, ra, rb);
  }

  // Writeback: Q/K as [B,H,S,64]; V transposed as [B,H,64,S]
#pragma unroll
  for (int nt = 0; nt < 4; ++nt) {
#pragma unroll
    for (int r = 0; r < 8; ++r) {
      int row = rowBase + wid * 16 + r + 8 * half;
      int col = colBase + nt * 16 + (lane & 15);
      float v = acc[nt][r] + bias[col];
      int b = row >> 11, s = row & (SEQ - 1);
      int h = col >> 6, d = col & 63;
      if (z < 2) {
        bf16* o = (z == 0) ? Q : Ko;
        o[(((size_t)b * NUM_HEADS + h) * SEQ + s) * DEPTH + d] = (bf16)v;
      } else {
        Vt[(((size_t)b * NUM_HEADS + h) * DEPTH + d) * SEQ + s] = (bf16)v;
      }
    }
  }
}

// ---------------- Flash attention ----------------
// grid = (SEQ/(16*AW), B*H), block = 32*AW. Each block shares K/V tiles via LDS
// across AW waves (8x L2 traffic reduction); each wave owns a 16-query tile.
__global__ __launch_bounds__(32 * AW)
void k_attn(const bf16* __restrict__ Q, const bf16* __restrict__ K,
            const bf16* __restrict__ Vt, bf16* __restrict__ Cc) {
  __shared__ __align__(16) bf16 Ks[64][72];       // [key][d]
  __shared__ __align__(16) bf16 Vs[64][72];       // [d][key]
  __shared__ __align__(16) bf16 Pl[AW][16][72];   // per-wave P tile [row][key]
  const int bh = blockIdx.y;
  const int b = bh >> 4, h = bh & 15;
  const int t = threadIdx.x;
  const int wid = t >> 5;
  const int lane = t & 31;
  const int half = lane >> 4;
  const int qbase = (blockIdx.x * AW + wid) * 16;
  const bf16* Qp = Q + ((size_t)bh * SEQ + qbase) * DEPTH;
  const bf16* Kp = K + (size_t)bh * SEQ * DEPTH;
  const bf16* Vp = Vt + (size_t)bh * DEPTH * SEQ;

  v16bf qf0 = load_frag_a(Qp, DEPTH);       // d 0..31
  v16bf qf1 = load_frag_a(Qp + 32, DEPTH);  // d 32..63

  v8f acc[4] = {};
  float mrow[8], lrow[8];
#pragma unroll
  for (int r = 0; r < 8; ++r) { mrow[r] = -1e30f; lrow[r] = 0.f; }

  for (int kt = 0; kt < SEQ; kt += 64) {
    __syncthreads();  // prev iter's LDS reads done before overwrite
    // cooperative load: K tile [64 keys][64 d], V tile [64 d][64 keys]
#pragma unroll
    for (int i = 0; i < 4; ++i) {
      int idx = i * 256 + t;     // 0..1023, 4 bf16 each
      int r = idx >> 4;          // 0..63
      int c = (idx & 15) * 4;    // 0..60
      *(uint2*)&Ks[r][c] = *(const uint2*)&Kp[(size_t)(kt + r) * DEPTH + c];
      *(uint2*)&Vs[r][c] = *(const uint2*)&Vp[(size_t)r * SEQ + kt + c];
    }
    __syncthreads();

    // logits: 4 key-subtiles of 16
    v8f s[4];
#pragma unroll
    for (int j = 0; j < 4; ++j) {
      v16bf kb0 = load_frag_b(&Ks[j * 16][0], 72);
      v16bf kb1 = load_frag_b(&Ks[j * 16][32], 72);
      v8f sj = {};
      sj = wmma_bf16(qf0, kb0, sj);
      sj = wmma_bf16(qf1, kb1, sj);
      s[j] = sj;
    }

    float alpha[8];
#pragma unroll
    for (int r = 0; r < 8; ++r) {
      float a0 = s[0][r] * 0.125f;  // 1/sqrt(64)
      float a1 = s[1][r] * 0.125f;
      float a2 = s[2][r] * 0.125f;
      float a3 = s[3][r] * 0.125f;
      float mx = fmaxf(fmaxf(a0, a1), fmaxf(a2, a3));
#pragma unroll
      for (int off = 1; off < 16; off <<= 1)
        mx = fmaxf(mx, __shfl_xor(mx, off, 32));
      float mnew = fmaxf(mrow[r], mx);
      alpha[r] = __expf(mrow[r] - mnew);
      float p0 = __expf(a0 - mnew);
      float p1 = __expf(a1 - mnew);
      float p2 = __expf(a2 - mnew);
      float p3 = __expf(a3 - mnew);
      float ps = (p0 + p1) + (p2 + p3);
#pragma unroll
      for (int off = 1; off < 16; off <<= 1)
        ps += __shfl_xor(ps, off, 32);
      lrow[r] = lrow[r] * alpha[r] + ps;
      mrow[r] = mnew;
      s[0][r] = p0; s[1][r] = p1; s[2][r] = p2; s[3][r] = p3;
    }

    // Reshape P (C-layout) -> A-fragment via per-wave LDS tile
#pragma unroll
    for (int j = 0; j < 4; ++j)
#pragma unroll
      for (int r = 0; r < 8; ++r)
        Pl[wid][r + 8 * half][j * 16 + (lane & 15)] = (bf16)s[j][r];
    __syncthreads();
    v16bf pf0 = load_frag_a(&Pl[wid][0][0], 72);   // keys 0..31 of tile
    v16bf pf1 = load_frag_a(&Pl[wid][0][32], 72);  // keys 32..63

#pragma unroll
    for (int nt = 0; nt < 4; ++nt) {
      v16bf vf0 = load_frag_b(&Vs[nt * 16][0], 72);
      v16bf vf1 = load_frag_b(&Vs[nt * 16][32], 72);
      v8f c = acc[nt];
#pragma unroll
      for (int r = 0; r < 8; ++r) c[r] *= alpha[r];
      c = wmma_bf16(pf0, vf0, c);
      acc[nt] = wmma_bf16(pf1, vf1, c);
    }
  }

  // Normalize and write concat [B,S,1024]
#pragma unroll
  for (int nt = 0; nt < 4; ++nt) {
#pragma unroll
    for (int r = 0; r < 8; ++r) {
      int s = qbase + r + 8 * half;
      int col = h * DEPTH + nt * 16 + (lane & 15);
      Cc[((size_t)b * SEQ + s) * D_MODEL + col] = (bf16)(acc[nt][r] / lrow[r]);
    }
  }
}

// ---------------- Output projection: concat @ wo + bo -> f32 ----------------
__global__ __launch_bounds__(256)
void k_gemm_out(const bf16* __restrict__ X, const bf16* __restrict__ W,
                const float* __restrict__ bias, float* __restrict__ Out) {
  __shared__ __align__(16) bf16 As[2][BM][LDT];
  __shared__ __align__(16) bf16 Bs[2][BN][LDT];
  const int rowBase = blockIdx.y * BM;
  const int colBase = blockIdx.x * BN;
  const int t = threadIdx.x;
  const int wid = t >> 5;
  const int lane = t & 31;
  const int half = lane >> 4;

  v8f acc[4] = {};
  unsigned int ra[8], rb[4];

  gemm_stage(X, W, rowBase, colBase, 0, t, ra, rb);
  gemm_commit(As[0], Bs[0], t, ra, rb);

  const int NIT = D_MODEL / BK;
  for (int it = 0; it < NIT; ++it) {
    const int buf = it & 1;
    __syncthreads();
    const bool pre = (it + 1 < NIT);
    if (pre) gemm_stage(X, W, rowBase, colBase, (it + 1) * BK, t, ra, rb);

    v16bf a = load_frag_a(&As[buf][wid * 16][0], LDT);
    v16bf b0 = load_frag_b(&Bs[buf][0][0], LDT);
    v16bf b1 = load_frag_b(&Bs[buf][16][0], LDT);
    v16bf b2 = load_frag_b(&Bs[buf][32][0], LDT);
    v16bf b3 = load_frag_b(&Bs[buf][48][0], LDT);
    acc[0] = wmma_bf16(a, b0, acc[0]);
    acc[1] = wmma_bf16(a, b1, acc[1]);
    acc[2] = wmma_bf16(a, b2, acc[2]);
    acc[3] = wmma_bf16(a, b3, acc[3]);

    if (pre) gemm_commit(As[buf ^ 1], Bs[buf ^ 1], t, ra, rb);
  }

#pragma unroll
  for (int nt = 0; nt < 4; ++nt) {
#pragma unroll
    for (int r = 0; r < 8; ++r) {
      int row = rowBase + wid * 16 + r + 8 * half;
      int col = colBase + nt * 16 + (lane & 15);
      Out[(size_t)row * D_MODEL + col] = acc[nt][r] + bias[col];
    }
  }
}

extern "C" void kernel_launch(void* const* d_in, const int* in_sizes, int n_in,
                              void* d_out, int out_size, void* d_ws, size_t ws_size,
                              hipStream_t stream) {
  const float* x  = (const float*)d_in[0];
  const float* wq = (const float*)d_in[1];
  const float* bq = (const float*)d_in[2];
  const float* wk = (const float*)d_in[3];
  const float* bk = (const float*)d_in[4];
  const float* wv = (const float*)d_in[5];
  const float* bv = (const float*)d_in[6];
  const float* wo = (const float*)d_in[7];
  const float* bo = (const float*)d_in[8];

  char* ws = (char*)d_ws;
  bf16* xb  = (bf16*)(ws);                       // 8 MB
  bf16* wqb = (bf16*)(ws + (8ull  << 20));       // 2 MB each
  bf16* wkb = (bf16*)(ws + (10ull << 20));
  bf16* wvb = (bf16*)(ws + (12ull << 20));
  bf16* wob = (bf16*)(ws + (14ull << 20));
  bf16* Qb  = (bf16*)(ws + (16ull << 20));       // 8 MB [B,H,S,64]
  bf16* Kb  = (bf16*)(ws + (24ull << 20));       // 8 MB [B,H,S,64]
  bf16* Vtb = (bf16*)(ws + (32ull << 20));       // 8 MB [B,H,64,S]
  bf16* Cc  = (bf16*)(ws + (40ull << 20));       // 8 MB [B,S,1024]

  const int nx = ROWS * D_MODEL;
  const int nw = D_MODEL * D_MODEL;
  k_cvt<<<(nx + 255) / 256, 256, 0, stream>>>(x,  xb,  nx);
  k_cvt<<<(nw + 255) / 256, 256, 0, stream>>>(wq, wqb, nw);
  k_cvt<<<(nw + 255) / 256, 256, 0, stream>>>(wk, wkb, nw);
  k_cvt<<<(nw + 255) / 256, 256, 0, stream>>>(wv, wvb, nw);
  k_cvt<<<(nw + 255) / 256, 256, 0, stream>>>(wo, wob, nw);

  dim3 g1(D_MODEL / BN, ROWS / BM, 3);
  k_gemm_qkv<<<g1, 256, 0, stream>>>(xb, wqb, wkb, wvb, bq, bk, bv, Qb, Kb, Vtb);

  dim3 g2(SEQ / (16 * AW), BATCH * NUM_HEADS, 1);
  k_attn<<<g2, 32 * AW, 0, stream>>>(Qb, Kb, Vtb, Cc);

  dim3 g3(D_MODEL / BN, ROWS / BM, 1);
  k_gemm_out<<<g3, 256, 0, stream>>>(Cc, wob, bo, (float*)d_out);
}
